// Linear_mvm_82016695484431
// MI455X (gfx1250) — 8-bit matrix operand order repaired and checked by reading, compiled, not run on hardware
//
#include <hip/hip_runtime.h>
#include <hip/hip_bf16.h>

typedef __attribute__((ext_vector_type(8))) int v8i;

#define IN_F   1024
#define OUT_F  1024
#define BATCH  64
#define KSTEPS (IN_F / 64)

#define WS_XL  (0)
#define WS_XH  (BATCH * IN_F)
#define WS_WLP (2 * BATCH * IN_F)
#define WS_WHP (WS_WLP + OUT_F * IN_F)
#define WS_WLN (WS_WLP + 2 * OUT_F * IN_F)
#define WS_WHN (WS_WLP + 3 * OUT_F * IN_F)

__device__ __forceinline__ uint32_t pack4(int b0, int b1, int b2, int b3) {
    return (uint32_t)(b0 & 0xFF) | ((uint32_t)(b1 & 0xFF) << 8) |
           ((uint32_t)(b2 & 0xFF) << 16) | ((uint32_t)(b3 & 0xFF) << 24);
}

__device__ __forceinline__ int quant_x(float v) {
    float r = rintf(v * 4096.0f);
    r = fminf(fmaxf(r, -32768.0f), 32767.0f);
    return (int)r;
}

__device__ __forceinline__ unsigned quant_w(float v) {
    float r = rintf(fmaxf(v, 0.0f) * 4096.0f);
    r = fminf(r, 65535.0f);
    return (unsigned)r;
}

__device__ __forceinline__ int acm_quant(int acc) {
    int q   = acc >> 12;
    int rem = acc & 4095;
    if (rem > 2048 || (rem == 2048 && (q & 1))) q += 1;
    return min(max(q, -32768), 32767);
}

__global__ void __launch_bounds__(256)
quant_x_kernel(const float* __restrict__ x, uint8_t* __restrict__ ws) {
    const int t = blockIdx.x * 256 + threadIdx.x;
    const float4 v = ((const float4*)x)[t];
    const int i0 = quant_x(v.x), i1 = quant_x(v.y), i2 = quant_x(v.z), i3 = quant_x(v.w);
    ((uint32_t*)(ws + WS_XL))[t] = pack4(i0, i1, i2, i3);
    ((uint32_t*)(ws + WS_XH))[t] = pack4(i0 >> 8, i1 >> 8, i2 >> 8, i3 >> 8);
}

__global__ void __launch_bounds__(256)
quant_w_kernel(const float* __restrict__ w, uint8_t* __restrict__ ws) {
    const int t = blockIdx.x * 256 + threadIdx.x;
    const float4 v = ((const float4*)w)[t];
    const unsigned p0 = quant_w(v.x),  p1 = quant_w(v.y),  p2 = quant_w(v.z),  p3 = quant_w(v.w);
    const unsigned n0 = quant_w(-v.x), n1 = quant_w(-v.y), n2 = quant_w(-v.z), n3 = quant_w(-v.w);
    ((uint32_t*)(ws + WS_WLP))[t] = pack4((int)p0, (int)p1, (int)p2, (int)p3);
    ((uint32_t*)(ws + WS_WHP))[t] = pack4((int)(p0 >> 8), (int)(p1 >> 8), (int)(p2 >> 8), (int)(p3 >> 8));
    ((uint32_t*)(ws + WS_WLN))[t] = pack4((int)n0, (int)n1, (int)n2, (int)n3);
    ((uint32_t*)(ws + WS_WHN))[t] = pack4((int)(n0 >> 8), (int)(n1 >> 8), (int)(n2 >> 8), (int)(n3 >> 8));
}

__global__ void __launch_bounds__(128, 1)
puma_mvm_wmma(const uint8_t* __restrict__ ws,
              const float* __restrict__ bias,
              float* __restrict__ out) {
    const int lane  = threadIdx.x & 31;
    const int mtile = threadIdx.x >> 5;
    const int ntile = blockIdx.x;
    const int lo    = lane & 15;
    const int hi    = lane >> 4;

    const int m = mtile * 16 + lo;
    const int n = ntile * 16 + lo;

    const uint8_t* __restrict__ xl  = ws + WS_XL  + (size_t)m * IN_F;
    const uint8_t* __restrict__ xh  = ws + WS_XH  + (size_t)m * IN_F;
    const uint8_t* __restrict__ wlp = ws + WS_WLP + (size_t)n * IN_F;
    const uint8_t* __restrict__ whp = ws + WS_WHP + (size_t)n * IN_F;
    const uint8_t* __restrict__ wln = ws + WS_WLN + (size_t)n * IN_F;
    const uint8_t* __restrict__ whn = ws + WS_WHN + (size_t)n * IN_F;

    v8i aPhh = (v8i)0, aPhl = (v8i)0, aPlh = (v8i)0, aPll = (v8i)0;
    v8i aNhh = (v8i)0, aNhl = (v8i)0, aNlh = (v8i)0, aNll = (v8i)0;

    for (int ks = 0; ks < KSTEPS; ++ks) {
        const int kb = ks * 64;

        v8i aLo, aHi;
#pragma unroll
        for (int p = 0; p < 2; ++p) {
            const int off = kb + p * 32 + hi * 16;
            const uint4 tl = *(const uint4*)(xl + off);
            const uint4 th = *(const uint4*)(xh + off);
            aLo[4 * p] = (int)tl.x; aLo[4 * p + 1] = (int)tl.y; aLo[4 * p + 2] = (int)tl.z; aLo[4 * p + 3] = (int)tl.w;
            aHi[4 * p] = (int)th.x; aHi[4 * p + 1] = (int)th.y; aHi[4 * p + 2] = (int)th.z; aHi[4 * p + 3] = (int)th.w;
        }

        v8i bLp, bHp, bLn, bHn;
#pragma unroll
        for (int h = 0; h < 2; ++h) {
            const int off = kb + h * 32 + hi * 16;
            const uint4 t0 = *(const uint4*)(wlp + off);
            const uint4 t1 = *(const uint4*)(whp + off);
            const uint4 t2 = *(const uint4*)(wln + off);
            const uint4 t3 = *(const uint4*)(whn + off);
            bLp[4 * h] = (int)t0.x; bLp[4 * h + 1] = (int)t0.y; bLp[4 * h + 2] = (int)t0.z; bLp[4 * h + 3] = (int)t0.w;
            bHp[4 * h] = (int)t1.x; bHp[4 * h + 1] = (int)t1.y; bHp[4 * h + 2] = (int)t1.z; bHp[4 * h + 3] = (int)t1.w;
            bLn[4 * h] = (int)t2.x; bLn[4 * h + 1] = (int)t2.y; bLn[4 * h + 2] = (int)t2.z; bLn[4 * h + 3] = (int)t2.w;
            bHn[4 * h] = (int)t3.x; bHn[4 * h + 1] = (int)t3.y; bHn[4 * h + 2] = (int)t3.z; bHn[4 * h + 3] = (int)t3.w;
        }

        aPhh = __builtin_amdgcn_wmma_i32_16x16x64_iu8(true,  aHi, false, bHp, aPhh, false, false);
        aPhl = __builtin_amdgcn_wmma_i32_16x16x64_iu8(true,  aHi, false, bLp, aPhl, false, false);
        aPlh = __builtin_amdgcn_wmma_i32_16x16x64_iu8(false, aLo, false, bHp, aPlh, false, false);
        aPll = __builtin_amdgcn_wmma_i32_16x16x64_iu8(false, aLo, false, bLp, aPll, false, false);
        aNhh = __builtin_amdgcn_wmma_i32_16x16x64_iu8(true,  aHi, false, bHn, aNhh, false, false);
        aNhl = __builtin_amdgcn_wmma_i32_16x16x64_iu8(true,  aHi, false, bLn, aNhl, false, false);
        aNlh = __builtin_amdgcn_wmma_i32_16x16x64_iu8(false, aLo, false, bHn, aNlh, false, false);
        aNll = __builtin_amdgcn_wmma_i32_16x16x64_iu8(false, aLo, false, bLn, aNll, false, false);
    }

    const float bval = bias[n];
#pragma unroll
    for (int j = 0; j < 8; ++j) {
        const int accP = (int)(((unsigned)aPhh[j] << 16) +
                               ((unsigned)(aPhl[j] + aPlh[j]) << 8) +
                               (unsigned)aPll[j]);
        const int accN = (int)(((unsigned)aNhh[j] << 16) +
                               ((unsigned)(aNhl[j] + aNlh[j]) << 8) +
                               (unsigned)aNll[j]);
        const int qp = acm_quant(accP);
        const int qn = acm_quant(accN);
        const int mrow = mtile * 16 + j + hi * 8;
        out[(size_t)mrow * OUT_F + ntile * 16 + lo] =
            (float)(qp - qn) * (1.0f / 4096.0f) + bval;
    }
}

extern "C" void kernel_launch(void* const* d_in, const int* in_sizes, int n_in,
                              void* d_out, int out_size, void* d_ws, size_t ws_size,
                              hipStream_t stream) {
    (void)in_sizes; (void)n_in; (void)ws_size; (void)out_size;
    const float* x    = (const float*)d_in[0];
    const float* w    = (const float*)d_in[1];
    const float* bias = (const float*)d_in[2];
    float* out        = (float*)d_out;
    uint8_t* ws       = (uint8_t*)d_ws;

    quant_x_kernel<<<dim3((BATCH * IN_F / 4) / 256), dim3(256), 0, stream>>>(x, ws);
    quant_w_kernel<<<dim3((OUT_F * IN_F / 4) / 256), dim3(256), 0, stream>>>(w, ws);

    puma_mvm_wmma<<<dim3(OUT_F / 16), dim3(128), 0, stream>>>(ws, bias, out);
}
